// PNAConv_Module_4999341932621
// MI455X (gfx1250) — compile-verified, matrix operand order checked by reading, not run on hardware
//
#include <hip/hip_runtime.h>
#include <hip/hip_bf16.h>
#include <math.h>

// ---------------- problem constants ----------------
#define NN      30000          // nodes
#define NE      480000         // edges
#define F       256            // in_size
#define EF      32             // edge feat
#define T       4              // towers
#define DD      64             // tower dim
#define M_IN    160            // 2*DD + EF
#define U_IN    832            // 13*DD
#define DELTA   2.5f

typedef __attribute__((ext_vector_type(16))) __bf16 bf16x16;
typedef __attribute__((ext_vector_type(8)))  __bf16 bf16x8;
typedef __attribute__((ext_vector_type(8)))  float  f32x8;

// ---------------- fragment builders ----------------
// 16-bit A/B fragment layout (ISA 7.12.2): element j holds
// K = (j<8 ? kb+j : 16+kb+(j-8)), kb = 8*(lane>>4).
// Caller passes lo = base+kb, hi = base+16+kb; both 16B-aligned ->
// each run is a single global_load_b128 of 8 bf16, no cvt needed.
__device__ __forceinline__ bf16x16 frag_ldb(const __bf16* lo, const __bf16* hi) {
    bf16x8 a = *reinterpret_cast<const bf16x8*>(lo);
    bf16x8 b = *reinterpret_cast<const bf16x8*>(hi);
    bf16x16 r;
    #pragma unroll
    for (int j = 0; j < 8; ++j) { r[j] = a[j]; r[8 + j] = b[j]; }
    return r;
}

__device__ __forceinline__ f32x8 wmma_bf16(bf16x16 a, bf16x16 b, f32x8 c) {
    return __builtin_amdgcn_wmma_f32_16x16x32_bf16(false, a, false, b, (short)0, c, false, false);
}

// Native CDNA5 float atomics (global_atomic_add_f32 / max_num_f32 / min_num_f32,
// non-returning, device scope).
__device__ __forceinline__ void atomAddF(float* a, float v) {
    __hip_atomic_fetch_add(a, v, __ATOMIC_RELAXED, __HIP_MEMORY_SCOPE_AGENT);
}
__device__ __forceinline__ void atomMaxF(float* a, float v) {
    __hip_atomic_fetch_max(a, v, __ATOMIC_RELAXED, __HIP_MEMORY_SCOPE_AGENT);
}
__device__ __forceinline__ void atomMinF(float* a, float v) {
    __hip_atomic_fetch_min(a, v, __ATOMIC_RELAXED, __HIP_MEMORY_SCOPE_AGENT);
}

// ---------------- prep: f32 -> bf16 mirror ----------------
__global__ void k_cvt(const float* __restrict__ in, __bf16* __restrict__ out, long n) {
    long i = (long)blockIdx.x * blockDim.x + threadIdx.x;
    if (i < n) out[i] = (__bf16)in[i];
}

// ---------------- init accumulators ----------------
__global__ void k_init(float* sumB, float* ssqB, float* maxB, float* minB, float* degB) {
    size_t i = (size_t)blockIdx.x * blockDim.x + threadIdx.x;
    if (i < (size_t)NN * F) {
        sumB[i] = 0.0f;
        ssqB[i] = 0.0f;
        maxB[i] = __uint_as_float(0xFF800000u);   // -inf
        minB[i] = __uint_as_float(0x7F800000u);   // +inf
    }
    if (i < (size_t)NN) degB[i] = 0.0f;
}

// ---------------- degree ----------------
__global__ void k_deg(const int* __restrict__ dst, float* degB) {
    int e = blockIdx.x * blockDim.x + threadIdx.x;
    if (e < NE) atomAddF(degB + dst[e], 1.0f);
}

// ---------------- edge message WMMA + fused scatter-reduce ----------------
// 1 wave = 16 edges; msg(16x64) = f(16x160 bf16) x Mw[t]^T per tower.
__global__ void k_edge(const __bf16* __restrict__ nfB, const __bf16* __restrict__ efB,
                       const int* __restrict__ src, const int* __restrict__ dst,
                       const __bf16* __restrict__ MwB, const float* __restrict__ Mb,
                       float* sumB, float* ssqB, float* maxB, float* minB) {
    int wave = (blockIdx.x * blockDim.x + threadIdx.x) >> 5;
    int lane = threadIdx.x & 31;
    int nTiles = (NE + 15) / 16;
    if (wave >= nTiles) return;                     // wave-uniform exit

    int row  = lane & 15;
    int half = lane >> 4;
    int kb   = half * 8;

    int eA  = wave * 16 + row;                      // A-matrix row edge for this lane
    int eAc = eA < NE ? eA : 0;
    int es  = src[eAc];
    int edn = dst[eAc];

    // destination nodes of the 8 D-rows this lane owns (M = r + 8*half)
    int  drow[8];
    bool dval[8];
    #pragma unroll
    for (int r = 0; r < 8; ++r) {
        int ee  = wave * 16 + half * 8 + r;
        dval[r] = ee < NE;
        drow[r] = dst[dval[r] ? ee : 0];
    }

    const __bf16* hs = nfB + (size_t)es  * F;
    const __bf16* hd = nfB + (size_t)edn * F;
    const __bf16* fe = efB + (size_t)eAc * EF;
    bf16x16 A4 = frag_ldb(fe + kb, fe + 16 + kb);   // e_feat chunk, tower-invariant

    for (int t = 0; t < T; ++t) {
        const __bf16* hst = hs + t * DD;
        const __bf16* hdt = hd + t * DD;
        // K-chunks: [h_src(0..63) | h_dst(64..127) | e_feat(128..159)]
        bf16x16 A0 = frag_ldb(hst + kb,      hst + 16 + kb);
        bf16x16 A1 = frag_ldb(hst + 32 + kb, hst + 48 + kb);
        bf16x16 A2 = frag_ldb(hdt + kb,      hdt + 16 + kb);
        bf16x16 A3 = frag_ldb(hdt + 32 + kb, hdt + 48 + kb);

        for (int nt = 0; nt < 4; ++nt) {
            int o = nt * 16 + row;                  // output column (per tower)
            float bias = Mb[t * DD + o];
            f32x8 c;
            #pragma unroll
            for (int r = 0; r < 8; ++r) c[r] = bias;

            const __bf16* wr = MwB + ((size_t)(t * DD + o)) * M_IN;  // B[K][o] = Mw[t][o][K]
            c = wmma_bf16(A0, frag_ldb(wr       + kb, wr + 16  + kb), c);
            c = wmma_bf16(A1, frag_ldb(wr + 32  + kb, wr + 48  + kb), c);
            c = wmma_bf16(A2, frag_ldb(wr + 64  + kb, wr + 80  + kb), c);
            c = wmma_bf16(A3, frag_ldb(wr + 96  + kb, wr + 112 + kb), c);
            c = wmma_bf16(A4, frag_ldb(wr + 128 + kb, wr + 144 + kb), c);

            int col = t * DD + o;
            #pragma unroll
            for (int r = 0; r < 8; ++r) {
                if (!dval[r]) continue;
                float v = c[r];
                size_t idx = (size_t)drow[r] * F + col;
                atomAddF(sumB + idx, v);
                atomAddF(ssqB + idx, v * v);
                atomMaxF(maxB + idx, v);
                atomMinF(minB + idx, v);
            }
        }
    }
}

// ---------------- finalize aggregates -> bf16 agg tensor + scalers ----------------
// aggB layout per node: [t][mean(64)|max(64)|min(64)|std(64)]  -> u_in K-order.
__global__ void k_finalize(const float* __restrict__ sumB, const float* __restrict__ ssqB,
                           const float* __restrict__ maxB, const float* __restrict__ minB,
                           const float* __restrict__ degB,
                           __bf16* __restrict__ aggB, float* ampB, float* attB) {
    size_t i = (size_t)blockIdx.x * blockDim.x + threadIdx.x;
    if (i >= (size_t)NN * F) return;
    int n = (int)(i >> 8);
    int c = (int)(i & 255);
    float deg  = degB[n];
    float degc = fmaxf(deg, 1.0f);
    float mean = sumB[i] / degc;
    float var  = fmaxf(ssqB[i] / degc - mean * mean, 0.0f);
    float stdv = sqrtf(var + 1e-30f);
    bool  has  = deg > 0.0f;

    int t   = c >> 6;
    int idx = c & 63;
    size_t base = (size_t)n * (4 * DD * T / 1) * 1;  // n * 1024
    base = (size_t)n * 1024 + (size_t)t * 256 + idx;
    aggB[base]       = (__bf16)mean;
    aggB[base + 64]  = (__bf16)(has ? maxB[i] : 0.0f);
    aggB[base + 128] = (__bf16)(has ? minB[i] : 0.0f);
    aggB[base + 192] = (__bf16)stdv;

    if (c == 0) {
        float logd = logf(deg + 1.0f);
        ampB[n] = logd / DELTA;
        attB[n] = has ? (DELTA / fmaxf(logd, 1e-12f)) : 0.0f;
    }
}

// ---------------- U GEMM with scaler folding ----------------
// u_in = [h | agg | agg*amp | agg*att]; since amp/att are per-node (per A-row)
// scalars on whole blocks:  u = (h,agg)@W_id + amp*(agg@W_amp) + att*(agg@W_att).
// Same bf16 A fragments feed three accumulators -> no runtime scaling in A.
__global__ void k_ugemm(const __bf16* __restrict__ nfB, const __bf16* __restrict__ aggB,
                        const __bf16* __restrict__ UwB, const float* __restrict__ Ub,
                        const float* __restrict__ ampB, const float* __restrict__ attB,
                        float* __restrict__ uB) {
    int wave = (blockIdx.x * blockDim.x + threadIdx.x) >> 5;
    int lane = threadIdx.x & 31;
    int nTiles = (NN + 15) / 16;
    if (wave >= nTiles) return;

    int row  = lane & 15;
    int half = lane >> 4;
    int kb   = half * 8;
    int node = wave * 16 + row;
    if (node >= NN) node = NN - 1;                  // NN % 16 == 0; safety only

    // amp/att for the 8 D-rows this lane owns
    float amp8[8], att8[8];
    #pragma unroll
    for (int r = 0; r < 8; ++r) {
        int nr = wave * 16 + half * 8 + r;
        if (nr >= NN) nr = NN - 1;
        amp8[r] = ampB[nr];
        att8[r] = attB[nr];
    }

    for (int t = 0; t < T; ++t) {
        const __bf16* hp = nfB  + (size_t)node * F    + t * DD;
        const __bf16* gp = aggB + (size_t)node * 1024 + t * 256;
        bf16x16 Ah0 = frag_ldb(hp + kb,      hp + 16 + kb);
        bf16x16 Ah1 = frag_ldb(hp + 32 + kb, hp + 48 + kb);
        bf16x16 Ag[8];
        #pragma unroll
        for (int q = 0; q < 8; ++q)
            Ag[q] = frag_ldb(gp + q * 32 + kb, gp + q * 32 + 16 + kb);

        for (int nt = 0; nt < 4; ++nt) {
            int o = nt * 16 + row;
            const __bf16* wr = UwB + (size_t)(t * DD + o) * U_IN;
            float bias = Ub[t * DD + o];
            f32x8 ci, ca, ct;
            #pragma unroll
            for (int r = 0; r < 8; ++r) { ci[r] = bias; ca[r] = 0.0f; ct[r] = 0.0f; }

            // h chunks (identity group, K-chunks 0..1)
            ci = wmma_bf16(Ah0, frag_ldb(wr + kb,      wr + 16 + kb), ci);
            ci = wmma_bf16(Ah1, frag_ldb(wr + 32 + kb, wr + 48 + kb), ci);
            // agg chunks against identity / amp / att weight slices
            #pragma unroll
            for (int q = 0; q < 8; ++q) {
                const __bf16* wi = wr + (2  + q) * 32;
                const __bf16* wa = wr + (10 + q) * 32;
                const __bf16* wt = wr + (18 + q) * 32;
                ci = wmma_bf16(Ag[q], frag_ldb(wi + kb, wi + 16 + kb), ci);
                ca = wmma_bf16(Ag[q], frag_ldb(wa + kb, wa + 16 + kb), ca);
                ct = wmma_bf16(Ag[q], frag_ldb(wt + kb, wt + 16 + kb), ct);
            }
            #pragma unroll
            for (int r = 0; r < 8; ++r) {
                int nr = wave * 16 + half * 8 + r;
                if (nr < NN)
                    uB[(size_t)nr * F + t * DD + o] =
                        ci[r] + amp8[r] * ca[r] + att8[r] * ct[r];
            }
        }
    }
}

// ---------------- BatchNorm channel stats -> folded scale/bias ----------------
__global__ void k_bnstats(const float* __restrict__ uB,
                          const float* __restrict__ gamma, const float* __restrict__ beta,
                          float* scaleC, float* biasC) {
    int c = blockIdx.x;                               // 256 channels
    float s = 0.0f, q = 0.0f;
    for (int n = threadIdx.x; n < NN; n += blockDim.x) {
        float v = uB[(size_t)n * F + c];
        s += v; q += v * v;
    }
    __shared__ float shs[256];
    __shared__ float shq[256];
    shs[threadIdx.x] = s; shq[threadIdx.x] = q;
    __syncthreads();
    for (int off = 128; off > 0; off >>= 1) {
        if (threadIdx.x < off) {
            shs[threadIdx.x] += shs[threadIdx.x + off];
            shq[threadIdx.x] += shq[threadIdx.x + off];
        }
        __syncthreads();
    }
    if (threadIdx.x == 0) {
        float mu  = shs[0] / (float)NN;
        float var = shq[0] / (float)NN - mu * mu;
        float is  = rsqrtf(var + 1e-5f);
        float g   = gamma[c] * is;
        scaleC[c] = g;
        biasC[c]  = beta[c] - mu * g;
    }
}

// ---------------- apply BN affine -> bf16 u_norm ----------------
__global__ void k_norm(const float* __restrict__ uB,
                       const float* __restrict__ scaleC, const float* __restrict__ biasC,
                       __bf16* __restrict__ unB) {
    size_t i = (size_t)blockIdx.x * blockDim.x + threadIdx.x;
    if (i >= (size_t)NN * F) return;
    int c = (int)(i & 255);
    unB[i] = (__bf16)fmaf(uB[i], scaleC[c], biasC[c]);
}

// ---------------- mix WMMA + LeakyReLU + residual + ReLU ----------------
// 1 wave = 16 nodes x 64 outputs (4 otiles reuse the cached A fragments).
__global__ void k_mix(const __bf16* __restrict__ unB, const __bf16* __restrict__ mixWB,
                      const float* __restrict__ mixBias,
                      const float* __restrict__ nf, float* __restrict__ out) {
    int task = (blockIdx.x * blockDim.x + threadIdx.x) >> 5;
    int lane = threadIdx.x & 31;
    int nTiles = (NN + 15) / 16;
    if (task >= nTiles * 4) return;

    int tile = task >> 2;
    int og   = task & 3;
    int row  = lane & 15;
    int half = lane >> 4;
    int kb   = half * 8;
    int node = tile * 16 + row;
    if (node >= NN) node = NN - 1;

    bf16x16 A[8];
    const __bf16* up = unB + (size_t)node * F;
    #pragma unroll
    for (int kc = 0; kc < 8; ++kc)
        A[kc] = frag_ldb(up + kc * 32 + kb, up + kc * 32 + 16 + kb);

    for (int j = 0; j < 4; ++j) {
        int o = (og * 4 + j) * 16 + row;
        const __bf16* wr = mixWB + (size_t)o * F;     // B[K][o] = mixW[o][K]
        f32x8 acc;
        #pragma unroll
        for (int r = 0; r < 8; ++r) acc[r] = 0.0f;
        #pragma unroll
        for (int kc = 0; kc < 8; ++kc)
            acc = wmma_bf16(A[kc], frag_ldb(wr + kc * 32 + kb, wr + kc * 32 + 16 + kb), acc);

        float mb = mixBias[o];
        #pragma unroll
        for (int r = 0; r < 8; ++r) {
            int nr = tile * 16 + half * 8 + r;
            if (nr >= NN) continue;
            float v = acc[r] + mb;
            v = (v > 0.0f) ? v : 0.01f * v;           // LeakyReLU
            v += nf[(size_t)nr * F + o];              // residual
            out[(size_t)nr * F + o] = fmaxf(v, 0.0f); // final ReLU
        }
    }
}

// ---------------- host launcher ----------------
static inline size_t alignup(size_t x) { return (x + 255) & ~(size_t)255; }

extern "C" void kernel_launch(void* const* d_in, const int* in_sizes, int n_in,
                              void* d_out, int out_size, void* d_ws, size_t ws_size,
                              hipStream_t stream) {
    const float* nf      = (const float*)d_in[0];
    const float* ef      = (const float*)d_in[1];
    const int*   src     = (const int*)  d_in[2];
    const int*   dst     = (const int*)  d_in[3];
    const float* Mw      = (const float*)d_in[4];
    const float* Mb      = (const float*)d_in[5];
    const float* Uw      = (const float*)d_in[6];
    const float* Ub      = (const float*)d_in[7];
    const float* gamma   = (const float*)d_in[8];
    const float* beta    = (const float*)d_in[9];
    const float* mixW    = (const float*)d_in[10];
    const float* mixBias = (const float*)d_in[11];
    float*       out     = (float*)d_out;

    // ---- workspace carve-up (256B aligned, ~278 MB total) ----
    char*  base = (char*)d_ws;
    size_t cur  = 0;
    size_t NC   = (size_t)NN * F;
    auto take = [&](size_t bytes) { char* p = base + cur; cur += alignup(bytes); return p; };

    float*  sumB   = (float*) take(NC * 4);
    float*  ssqB   = (float*) take(NC * 4);
    float*  maxB   = (float*) take(NC * 4);
    float*  minB   = (float*) take(NC * 4);
    float*  uB     = (float*) take(NC * 4);
    __bf16* aggB   = (__bf16*)take((size_t)NN * 1024 * 2);
    __bf16* nfB    = (__bf16*)take(NC * 2);
    __bf16* efB    = (__bf16*)take((size_t)NE * EF * 2);
    __bf16* unB    = (__bf16*)take(NC * 2);
    __bf16* MwB    = (__bf16*)take((size_t)T * DD * M_IN * 2);
    __bf16* UwB    = (__bf16*)take((size_t)T * DD * U_IN * 2);
    __bf16* mixWB  = (__bf16*)take((size_t)F * F * 2);
    float*  degB   = (float*) take((size_t)NN * 4);
    float*  ampB   = (float*) take((size_t)NN * 4);
    float*  attB   = (float*) take((size_t)NN * 4);
    float*  scaleC = (float*) take(F * 4);
    float*  biasC  = (float*) take(F * 4);

    const int BT = 256;                                // 8 waves / block
    auto gfor = [&](long n) { return (int)((n + BT - 1) / BT); };

    // bf16 mirrors of all GEMM operands (one pass each, bandwidth-trivial)
    k_cvt<<<gfor((long)NC),              BT, 0, stream>>>(nf,   nfB,   (long)NC);
    k_cvt<<<gfor((long)NE * EF),         BT, 0, stream>>>(ef,   efB,   (long)NE * EF);
    k_cvt<<<gfor((long)T * DD * M_IN),   BT, 0, stream>>>(Mw,   MwB,   (long)T * DD * M_IN);
    k_cvt<<<gfor((long)T * DD * U_IN),   BT, 0, stream>>>(Uw,   UwB,   (long)T * DD * U_IN);
    k_cvt<<<gfor((long)F * F),           BT, 0, stream>>>(mixW, mixWB, (long)F * F);

    k_init<<<gfor((long)NC), BT, 0, stream>>>(sumB, ssqB, maxB, minB, degB);
    k_deg<<<gfor(NE), BT, 0, stream>>>(dst, degB);

    int eTiles = (NE + 15) / 16;                       // 30000 wave-tiles
    k_edge<<<(eTiles + 7) / 8, BT, 0, stream>>>(nfB, efB, src, dst, MwB, Mb,
                                                sumB, ssqB, maxB, minB);

    k_finalize<<<gfor((long)NC), BT, 0, stream>>>(sumB, ssqB, maxB, minB, degB,
                                                  aggB, ampB, attB);

    int nTiles = (NN + 15) / 16;                       // 1875 wave-tiles
    k_ugemm<<<(nTiles + 7) / 8, BT, 0, stream>>>(nfB, aggB, UwB, Ub, ampB, attB, uB);

    k_bnstats<<<F, 256, 0, stream>>>(uB, gamma, beta, scaleC, biasC);
    k_norm<<<gfor((long)NC), BT, 0, stream>>>(uB, scaleC, biasC, unB);

    k_mix<<<(nTiles * 4 + 7) / 8, BT, 0, stream>>>(unB, mixWB, mixBias, nf, out);
}
